// PeriodicLoss_79259326480950
// MI455X (gfx1250) — matrix operand order repaired and checked by reading, not compiled, not run on hardware
//
#include <hip/hip_runtime.h>
#include <hip/hip_bf16.h>
#include <math.h>

// ---------------------------------------------------------------------------
// PeriodicLoss on MI455X (gfx1250, wave32)
// 1024-pt rFFT per row via four-step (32 x 32) factorization:
//   stage A : Y = D32 @ reshape(x,32,32)            (real input -> 2 GEMMs)
//   stage B : Z = Y * twiddle(k1,n2)                (pointwise, LDS table)
//   stage C : X^T = D32 @ Z^T                       (complex; only k2<16 tiles,
//                                                    bin 512 via direct sum)
// Each 32x32 GEMM tile = one v_wmma_f32_16x16x32_f16 (f16 in, f32 accumulate).
// Rows are double-buffered with GLOBAL_LOAD_ASYNC_TO_LDS (ASYNCcnt) so the
// HBM stream (the roofline limiter: 64 MiB @ 23.3 TB/s ~ 2.9 us) overlaps the
// WMMA/reduction phase of the previous row.
// ---------------------------------------------------------------------------

typedef __attribute__((ext_vector_type(16))) _Float16     v16h;
typedef __attribute__((ext_vector_type(8)))  float        v8f;
typedef __attribute__((ext_vector_type(8)))  unsigned int v8u;

#define NFFT            1024
#define NF              513
#define WAVES_PER_BLOCK 4
#define BLOCK_T         (WAVES_PER_BLOCK * 32)
#define ROWS_PER_WAVE   4

#if defined(__gfx1250__) && defined(__has_builtin)
#if __has_builtin(__builtin_amdgcn_global_load_async_to_lds_b128)
#define HAVE_ASYNC 1
#endif
#endif
#ifndef HAVE_ASYNC
#define HAVE_ASYNC 0
#endif

static __device__ __forceinline__ v8f wmma_f16(v16h a, v16h b, v8f c) {
  // 8 args: (neg_a, A, neg_b, B, c_mod, C, reuse_a, reuse_b)
  return __builtin_amdgcn_wmma_f32_16x16x32_f16(false, a, false, b, (short)0, c,
                                                false, false);
}

static __device__ __forceinline__ void wait_ds() {
  __asm__ volatile("s_wait_dscnt 0" ::: "memory");
}
static __device__ __forceinline__ void wait_async() {
#if HAVE_ASYNC
  __asm__ volatile("s_wait_asynccnt 0x0" ::: "memory");
#endif
}

#if HAVE_ASYNC
// Builtin signature (from toolchain diagnostic): param0 = AS(1) int4*,
// param1 = AS(3) int4*, then imm offset and cpol.  Keep offset a literal 0
// (must be an ICE) and put the per-chunk offset in the pointers.
typedef int v4i __attribute__((vector_size(16)));
typedef __attribute__((address_space(1))) v4i* gv4i_p;
typedef __attribute__((address_space(3))) v4i* lv4i_p;

// Issue an async DMA of one 4KB row (f32[1024]) global -> LDS staging.
// Each lane moves 16B x 8.
static __device__ __forceinline__ void prefetch_row(const float* gsrc,
                                                    float* lds_dst, int lane) {
  const char* g = (const char*)gsrc + lane * 16;
  char*       l = (char*)lds_dst + lane * 16;
#pragma unroll
  for (int i = 0; i < 8; ++i) {
    __builtin_amdgcn_global_load_async_to_lds_b128(
        (gv4i_p)(g + i * 512), (lv4i_p)(l + i * 512), 0, 0);
  }
}
#endif

__global__ void __launch_bounds__(BLOCK_T)
periodic_main(const float* __restrict__ preds, const int* __restrict__ pFs,
              float* __restrict__ ws, int B, int nWaves) {
  __shared__ float2 tw[1024];                    // twiddle e^{-i 2pi k1 n2/1024}
  __shared__ float  accum[544];                  // block-level batch PSD
  __shared__ __align__(16) unsigned char wrk[WAVES_PER_BLOCK][4096];
  __shared__ __align__(16) float stg[WAVES_PER_BLOCK][1024];  // async staging

  const int lane = threadIdx.x & 31;
  const int wv   = threadIdx.x >> 5;
  const int Fs   = pFs[0];

  // --- derived band constants (matches argmin on uniform freq grid) ---------
  const float step = (float)Fs / 1024.0f;                 // (Fs/2)/512
  int left  = (int)lroundf((40.0f / 60.0f) / step);
  int right = (int)lroundf(3.0f / step);
  left  = min(max(left, 0), NF - 1);
  right = min(max(right, left + 1), NF);
  int delta = max(1, (int)lroundf(0.1f * 2.0f * (float)NF / (float)Fs));

  // --- block init: twiddle table + batch accumulator ------------------------
  for (int idx = threadIdx.x; idx < 1024; idx += BLOCK_T) {
    int k1 = idx >> 5, n2 = idx & 31;
    float th = (float)(k1 * n2) * (6.28318530717958647692f / 1024.0f);
    float s, c;
    sincosf(th, &s, &c);
    tw[idx] = make_float2(c, s);
  }
  for (int idx = threadIdx.x; idx < 544; idx += BLOCK_T) accum[idx] = 0.0f;
  __syncthreads();

  // --- per-lane constant A operands (16x32 f16 DFT matrices) ----------------
  // A layout: lanes 0-15 M=0..15 K=0..15; lanes 16-31 M=0..15 K=16..31
  const int Mlo   = lane & 15;
  const int kbase = (lane & 16) ? 16 : 0;
  v16h C0, C1, S0, S1, P0;   // C=cos, S=-sin, P=+sin  (rows M and M+16)
#pragma unroll
  for (int v = 0; v < 8; ++v) {
#pragma unroll
    for (int h = 0; h < 2; ++h) {
      int K = kbase + 2 * v + h;
      int e = 2 * v + h;
      float a0 = (float)((Mlo * K) & 31) * (3.14159265358979f / 16.0f);
      float s0, c0;
      sincosf(a0, &s0, &c0);
      C0[e] = (_Float16)c0;  S0[e] = (_Float16)(-s0);  P0[e] = (_Float16)s0;
      float a1 = (float)(((Mlo + 16) * K) & 31) * (3.14159265358979f / 16.0f);
      float s1, c1;
      sincosf(a1, &s1, &c1);
      C1[e] = (_Float16)c1;  S1[e] = (_Float16)(-s1);
    }
  }

  _Float16* raw  = (_Float16*)wrk[wv];          // [0..1023]  rawT / zre / psd
  _Float16* zim  = (_Float16*)wrk[wv] + 1024;   // [1024..2047] zim
  float*    psd  = (float*)wrk[wv];             // overlays raw+zim (dead then)
  const v8f vzero = {};

  const int gw = blockIdx.x * WAVES_PER_BLOCK + wv;

#if HAVE_ASYNC
  if (gw < B) prefetch_row(preds + (size_t)gw * NFFT, stg[wv], lane);
#endif

  for (int row = gw; row < B; row += nWaves) {
    // ---- obtain row, convert to f16, store transposed rawT[32*n2 + n1] ----
    float a512 = 0.0f;   // sum (-1)^n x[n]  -> X[512] (real)
#if HAVE_ASYNC
    wait_async();        // staging buffer for this row is complete
    const float4* rp = (const float4*)stg[wv];
#else
    const float4* rp = (const float4*)(preds + (size_t)row * NFFT);
#endif
#pragma unroll
    for (int i = 0; i < 8; ++i) {
      float4 x = rp[lane + 32 * i];
      int n0 = 4 * (lane + 32 * i);
      raw[((n0 + 0) & 31) * 32 + ((n0 + 0) >> 5)] = (_Float16)x.x;
      raw[((n0 + 1) & 31) * 32 + ((n0 + 1) >> 5)] = (_Float16)x.y;
      raw[((n0 + 2) & 31) * 32 + ((n0 + 2) >> 5)] = (_Float16)x.z;
      raw[((n0 + 3) & 31) * 32 + ((n0 + 3) >> 5)] = (_Float16)x.w;
      a512 += x.x - x.y + x.z - x.w;
    }
    wait_ds();           // staging reads + rawT writes drained

#if HAVE_ASYNC
    // Prefetch next row; DMA overlaps the whole WMMA/reduction phase below.
    {
      int nxt = row + nWaves;
      if (nxt < B) prefetch_row(preds + (size_t)nxt * NFFT, stg[wv], lane);
    }
#endif

    // ---- stage A: Y = D32 @ M  (8 WMMA) -----------------------------------
    v16h Ba0, Ba1;
    {
      int N0 = (lane & 15), kh = (lane & 16) ? 16 : 0;
      const unsigned int* p0 = (const unsigned int*)(raw + 32 * N0 + kh);
      const unsigned int* p1 = (const unsigned int*)(raw + 32 * (N0 + 16) + kh);
      v8u u0, u1;
#pragma unroll
      for (int v = 0; v < 8; ++v) { u0[v] = p0[v]; u1[v] = p1[v]; }
      Ba0 = __builtin_bit_cast(v16h, u0);
      Ba1 = __builtin_bit_cast(v16h, u1);
    }
    v8f yre00 = wmma_f16(C0, Ba0, vzero);
    v8f yre01 = wmma_f16(C0, Ba1, vzero);
    v8f yre10 = wmma_f16(C1, Ba0, vzero);
    v8f yre11 = wmma_f16(C1, Ba1, vzero);
    v8f yim00 = wmma_f16(S0, Ba0, vzero);
    v8f yim01 = wmma_f16(S0, Ba1, vzero);
    v8f yim10 = wmma_f16(S1, Ba0, vzero);
    v8f yim11 = wmma_f16(S1, Ba1, vzero);

    // ---- stage B: twiddle + store Z row-major for stage-C B operands ------
    auto twz = [&](v8f yr, v8f yi, int m, int t) {
#pragma unroll
      for (int r = 0; r < 8; ++r) {
        int k1 = 16 * m + r + ((lane & 16) ? 8 : 0);
        int n2 = 16 * t + (lane & 15);
        float2 w2 = tw[(k1 << 5) + n2];
        float zr = yr[r] * w2.x + yi[r] * w2.y;
        float zi = yi[r] * w2.x - yr[r] * w2.y;
        raw[(k1 << 5) + n2] = (_Float16)zr;
        zim[(k1 << 5) + n2] = (_Float16)zi;
      }
    };
    twz(yre00, yim00, 0, 0);
    twz(yre01, yim01, 0, 1);
    twz(yre10, yim10, 1, 0);
    twz(yre11, yim11, 1, 1);
    wait_ds();

    // ---- stage C: X^T = D32 @ Z^T, only k2 = 0..15 (8 WMMA) ---------------
    v16h Br0, Br1, Bi0, Bi1;
    {
      int N0 = (lane & 15), kh = (lane & 16) ? 16 : 0;
      const unsigned int* pr0 = (const unsigned int*)(raw + 32 * N0 + kh);
      const unsigned int* pr1 = (const unsigned int*)(raw + 32 * (N0 + 16) + kh);
      const unsigned int* pi0 = (const unsigned int*)(zim + 32 * N0 + kh);
      const unsigned int* pi1 = (const unsigned int*)(zim + 32 * (N0 + 16) + kh);
      v8u ur0, ur1, ui0, ui1;
#pragma unroll
      for (int v = 0; v < 8; ++v) {
        ur0[v] = pr0[v]; ur1[v] = pr1[v]; ui0[v] = pi0[v]; ui1[v] = pi1[v];
      }
      Br0 = __builtin_bit_cast(v16h, ur0);
      Br1 = __builtin_bit_cast(v16h, ur1);
      Bi0 = __builtin_bit_cast(v16h, ui0);
      Bi1 = __builtin_bit_cast(v16h, ui1);
    }
    v8f xr0 = wmma_f16(C0, Br0, vzero);  xr0 = wmma_f16(P0, Bi0, xr0);
    v8f xi0 = wmma_f16(C0, Bi0, vzero);  xi0 = wmma_f16(S0, Br0, xi0);
    v8f xr1 = wmma_f16(C0, Br1, vzero);  xr1 = wmma_f16(P0, Bi1, xr1);
    v8f xi1 = wmma_f16(C0, Bi1, vzero);  xi1 = wmma_f16(S0, Br1, xi1);

    // ---- PSD -> LDS (bins 0..511), bin 512 from alternating sum -----------
    auto pst = [&](v8f xr, v8f xi, int t) {
#pragma unroll
      for (int r = 0; r < 8; ++r) {
        int k2 = r + ((lane & 16) ? 8 : 0);
        int k1 = 16 * t + (lane & 15);
        float pr = xr[r], pi = xi[r];
        psd[k1 + (k2 << 5)] = pr * pr + pi * pi;
      }
    };
    pst(xr0, xi0, 0);
    pst(xr1, xi1, 1);
#pragma unroll
    for (int off = 16; off; off >>= 1) a512 += __shfl_xor(a512, off, 32);
    if (lane == 0) psd[512] = a512 * a512;
    wait_ds();

    // ---- per-row reductions -----------------------------------------------
    float tot = 0.f, ob = 0.f, band = 0.f, bmax = -1e30f, wsum = 0.f;
    int bidx = left;
    for (int k = lane; k < NF; k += 32) {
      float p = psd[k];
      tot += p;
      if (k < left || k >= right) {
        ob += p;
      } else {
        band += p;
        if (p > bmax) { bmax = p; bidx = k; }
      }
    }
#pragma unroll
    for (int off = 16; off; off >>= 1) {
      tot  += __shfl_xor(tot, off, 32);
      ob   += __shfl_xor(ob, off, 32);
      band += __shfl_xor(band, off, 32);
      float ov = __shfl_xor(bmax, off, 32);
      int   oi = __shfl_xor(bidx, off, 32);
      if (ov > bmax || (ov == bmax && oi < bidx)) { bmax = ov; bidx = oi; }
    }
    // in-band energy inside +-delta window of the peak
    for (int kk = bidx - delta + lane; kk < bidx + delta; kk += 32)
      if (kk >= left && kk < right) wsum += psd[kk];
#pragma unroll
    for (int off = 16; off; off >>= 1) wsum += __shfl_xor(wsum, off, 32);

    if (lane == 0) {
      unsafeAtomicAdd(&ws[0], ob / (1e-8f + tot));
      float den = band + 1e-8f * (float)(right - left);
      unsafeAtomicAdd(&ws[1], (band - wsum) / den);
    }
    // batch PSD accumulation into block-shared buffer
    for (int k = lane; k < NF; k += 32) unsafeAtomicAdd(&accum[k], psd[k]);
  }

  __syncthreads();
  for (int t = threadIdx.x; t < NF; t += BLOCK_T)
    unsafeAtomicAdd(&ws[2 + t], accum[t]);
}

// ---------------------------------------------------------------------------
__global__ void zero_ws(float* ws, int n) {
  int i = blockIdx.x * blockDim.x + threadIdx.x;
  if (i < n) ws[i] = 0.0f;
}

__global__ void __launch_bounds__(1024)
finalize_kernel(const float* __restrict__ ws, float* __restrict__ out, int B) {
  __shared__ float q[1024];
  __shared__ float red[1024];
  const int t = threadIdx.x;

  float v = (t < NF) ? ws[2 + t] : 0.0f;
  red[t] = v;
  __syncthreads();
  for (int off = 512; off > 0; off >>= 1) {
    if (t < off) red[t] += red[t + off];
    __syncthreads();
  }
  const float total = red[0];
  __syncthreads();

  // inclusive scan of normalized batch PSD (Hillis-Steele)
  q[t] = v / (1e-8f + total);
  __syncthreads();
  for (int off = 1; off < 1024; off <<= 1) {
    float add = (t >= off) ? q[t - off] : 0.0f;
    __syncthreads();
    q[t] += add;
    __syncthreads();
  }
  float Q = fminf(fmaxf(q[t], 0.0f), 1.0f);
  float P = fminf((float)t / 512.0f, 1.0f);
  float d = (t < NF) ? (P - Q) * (P - Q) : 0.0f;
  red[t] = d;
  __syncthreads();
  for (int off = 512; off > 0; off >>= 1) {
    if (t < off) red[t] += red[t + off];
    __syncthreads();
  }
  if (t == 0) {
    float invB = 1.0f / (float)B;
    out[0] = ws[0] * invB + ws[1] * invB + red[0] / (float)NF;
  }
}

// ---------------------------------------------------------------------------
extern "C" void kernel_launch(void* const* d_in, const int* in_sizes, int n_in,
                              void* d_out, int out_size, void* d_ws, size_t ws_size,
                              hipStream_t stream) {
  const float* preds = (const float*)d_in[0];
  const int*   Fs    = (const int*)d_in[1];
  float*       out   = (float*)d_out;
  float*       ws    = (float*)d_ws;

  const int B = in_sizes[0] / NFFT;

  zero_ws<<<3, 256, 0, stream>>>(ws, 544);

  int blocks = (B + WAVES_PER_BLOCK * ROWS_PER_WAVE - 1) /
               (WAVES_PER_BLOCK * ROWS_PER_WAVE);
  if (blocks < 1) blocks = 1;
  const int nWaves = blocks * WAVES_PER_BLOCK;

  periodic_main<<<blocks, BLOCK_T, 0, stream>>>(preds, Fs, ws, B, nWaves);
  finalize_kernel<<<1, 1024, 0, stream>>>(ws, out, B);
}
